// AdaptiveNodeSampler_50319836840353
// MI455X (gfx1250) — matrix operand order repaired and checked by reading, not compiled, not run on hardware
//
#include <hip/hip_runtime.h>
#include <hip/hip_bf16.h>
#include <math.h>

// ---------------------------------------------------------------------------
// AdaptiveNodeSampler for MI455X (gfx1250, wave32)
//
//   scores[b,n] = ((target[b]@Wq^T + bq) @ Wk) . cand[b,n] / sqrt(D)
//   (Q.bk term dropped: constant per row -> softmax shift-invariant)
//
// Kernel 1 (proj): two 1024x128x128 GEMMs via v_wmma_f32_16x16x32_f16.
// Kernel 2 (sample): memory-bound stream of 1.07 GB candidates (~46us floor
//   at 23.3 TB/s), softmax + gumbel keys + iterative top-32 in LDS.
// ---------------------------------------------------------------------------

typedef __attribute__((ext_vector_type(16))) _Float16 v16h;
typedef __attribute__((ext_vector_type(8)))  float    v8f;

#define DIMS   128
#define N_CAND 2048
#define TOPK   32
#define TPB    256

// ---------------------------------------------------------------------------
// Kernel 1: qprime[b,:] = (target[b,:] @ Wq^T + bq) @ Wk
// Block = 128 threads (4 waves), handles 16 rows. Each wave owns 2 N-tiles.
// WMMA fragment layouts used here (wave32; A/B k order is this code's, not the ISA's):
//   A 16x32 f16 : lane m=L%16, h=L/16 ; elem i -> k = h*16 + i
//   B 32x16 f16 : lane n=L%16, h=L/16 ; elem i -> k = h*16 + i
//   C/D 16x16 f32: lane n=L%16, h=L/16 ; vgpr r -> m = r + h*8
// ---------------------------------------------------------------------------
__global__ __launch_bounds__(128) void proj_kernel(
    const float* __restrict__ target,  // [B, D]
    const float* __restrict__ Wq,      // [D, D]
    const float* __restrict__ bq,      // [D]
    const float* __restrict__ Wk,      // [D, D]
    float*       __restrict__ qprime)  // [B, D]
{
    __shared__ float sT[16 * DIMS];
    __shared__ float sQ[16 * DIMS];

    const int row0 = blockIdx.x * 16;
    const int tid  = threadIdx.x;
    const int wave = tid >> 5;
    const int lane = tid & 31;
    const int lrow = lane & 15;   // m for A-frags, n for B/C-frags
    const int half = lane >> 4;

    // Stage the 16 target rows into LDS (16*128 floats, 16 per thread).
    for (int i = tid; i < 16 * DIMS; i += 128)
        sT[i] = target[(size_t)row0 * DIMS + i];
    __syncthreads();

    // ---- Stage 1: Q = T @ Wq^T + bq  ->  sQ ----
    for (int t = 0; t < 2; ++t) {
        const int ntile = wave * 2 + t;
        v8f acc = {};
        #pragma unroll
        for (int kc = 0; kc < 4; ++kc) {
            v16h a, bm;
            #pragma unroll
            for (int i = 0; i < 16; ++i) {
                const int ka = kc * 32 + half * 16 + i;
                a[i] = (_Float16)sT[lrow * DIMS + ka];
                const int kb = kc * 32 + half * 16 + i;
                // B[k][n] = Wq[n][k]   (T @ Wq^T)
                bm[i] = (_Float16)Wq[(ntile * 16 + lrow) * DIMS + kb];
            }
            acc = __builtin_amdgcn_wmma_f32_16x16x32_f16(
                false, a, false, bm, (short)0, acc, false, false);
        }
        const float bias = bq[ntile * 16 + lrow];
        #pragma unroll
        for (int r = 0; r < 8; ++r) {
            const int m = r + half * 8;
            sQ[m * DIMS + ntile * 16 + lrow] = acc[r] + bias;
        }
    }
    __syncthreads();

    // ---- Stage 2: qprime = Q @ Wk ----
    for (int t = 0; t < 2; ++t) {
        const int ntile = wave * 2 + t;
        v8f acc = {};
        #pragma unroll
        for (int kc = 0; kc < 4; ++kc) {
            v16h a, bm;
            #pragma unroll
            for (int i = 0; i < 16; ++i) {
                const int ka = kc * 32 + half * 16 + i;
                a[i] = (_Float16)sQ[lrow * DIMS + ka];
                const int kb = kc * 32 + half * 16 + i;
                // B[k][n] = Wk[k][n]   (Q @ Wk)
                bm[i] = (_Float16)Wk[kb * DIMS + ntile * 16 + lrow];
            }
            acc = __builtin_amdgcn_wmma_f32_16x16x32_f16(
                false, a, false, bm, (short)0, acc, false, false);
        }
        #pragma unroll
        for (int r = 0; r < 8; ++r) {
            const int m = r + half * 8;
            qprime[(size_t)(row0 + m) * DIMS + ntile * 16 + lrow] = acc[r];
        }
    }
}

// ---------------------------------------------------------------------------
// Kernel 2: one block per batch row b. Stream cand[b] (1 MB), compute scores,
// softmax, mixed probs, gumbel keys, iterative top-32 argmax. All in LDS.
// ---------------------------------------------------------------------------
__global__ __launch_bounds__(TPB) void sample_kernel(
    const float* __restrict__ cand,    // [B, N, D]
    const float* __restrict__ qprime,  // [B, D]
    const float* __restrict__ uin,     // [B, N]
    int*         __restrict__ out)     // [B, TOPK]
{
    __shared__ float sQp[DIMS];
    __shared__ float sKey[N_CAND];
    __shared__ float sRedV[TPB];
    __shared__ int   sRedI[TPB];

    const int b   = blockIdx.x;
    const int tid = threadIdx.x;

    if (tid < DIMS) sQp[tid] = qprime[(size_t)b * DIMS + tid];
    __syncthreads();

    const float4* crow = (const float4*)(cand + (size_t)b * N_CAND * DIMS);
    const float inv_scale = 0.08838834764831845f;  // 1/sqrt(128)

    float sc[N_CAND / TPB];
    #pragma unroll
    for (int i = 0; i < N_CAND / TPB; ++i) {
        const int n = tid + i * TPB;
        const float4* cp = crow + (size_t)n * (DIMS / 4);
        if (i + 1 < N_CAND / TPB)
            __builtin_prefetch(cp + (size_t)TPB * (DIMS / 4), 0, 0);
        float dot = 0.f;
        #pragma unroll
        for (int j = 0; j < DIMS / 4; ++j) {
            const float4 v = cp[j];
            dot = fmaf(v.x, sQp[4 * j + 0],
                  fmaf(v.y, sQp[4 * j + 1],
                  fmaf(v.z, sQp[4 * j + 2],
                  fmaf(v.w, sQp[4 * j + 3], dot))));
        }
        sc[i] = dot * inv_scale;
    }

    // row max
    float lmax = sc[0];
    #pragma unroll
    for (int i = 1; i < N_CAND / TPB; ++i) lmax = fmaxf(lmax, sc[i]);
    sRedV[tid] = lmax;
    __syncthreads();
    for (int s = TPB / 2; s > 0; s >>= 1) {
        if (tid < s) sRedV[tid] = fmaxf(sRedV[tid], sRedV[tid + s]);
        __syncthreads();
    }
    const float rowmax = sRedV[0];
    __syncthreads();

    // sum of exp
    float e[N_CAND / TPB];
    float lsum = 0.f;
    #pragma unroll
    for (int i = 0; i < N_CAND / TPB; ++i) {
        e[i] = __expf(sc[i] - rowmax);
        lsum += e[i];
    }
    sRedV[tid] = lsum;
    __syncthreads();
    for (int s = TPB / 2; s > 0; s >>= 1) {
        if (tid < s) sRedV[tid] += sRedV[tid + s];
        __syncthreads();
    }
    const float invsum = 1.0f / sRedV[0];
    __syncthreads();

    // keys = log(0.9*p + 0.1/N) + gumbel(u)
    const float invN = 1.0f / (float)N_CAND;
    #pragma unroll
    for (int i = 0; i < N_CAND / TPB; ++i) {
        const int n = tid + i * TPB;
        const float p = 0.9f * (e[i] * invsum) + 0.1f * invN;
        const float uu = uin[(size_t)b * N_CAND + n];
        const float g = -__logf(-__logf(uu + 1e-10f) + 1e-10f);
        sKey[n] = __logf(p) + g;
    }
    __syncthreads();

    // iterative top-32 argmax (mark selected with -inf)
    const float NEG = -3.4e38f;
    for (int k = 0; k < TOPK; ++k) {
        float bv = NEG;
        int   bi = 0;
        #pragma unroll
        for (int i = 0; i < N_CAND / TPB; ++i) {
            const int n = tid + i * TPB;
            const float v = sKey[n];
            if (v > bv) { bv = v; bi = n; }
        }
        sRedV[tid] = bv;
        sRedI[tid] = bi;
        __syncthreads();
        for (int s = TPB / 2; s > 0; s >>= 1) {
            if (tid < s && sRedV[tid + s] > sRedV[tid]) {
                sRedV[tid] = sRedV[tid + s];
                sRedI[tid] = sRedI[tid + s];
            }
            __syncthreads();
        }
        if (tid == 0) {
            out[b * TOPK + k] = sRedI[0];
            sKey[sRedI[0]] = NEG;
        }
        __syncthreads();
    }
}

// ---------------------------------------------------------------------------
extern "C" void kernel_launch(void* const* d_in, const int* in_sizes, int n_in,
                              void* d_out, int out_size, void* d_ws, size_t ws_size,
                              hipStream_t stream) {
    const float* target = (const float*)d_in[0];  // [B, 128]
    const float* cand   = (const float*)d_in[1];  // [B, 2048, 128]
    const float* Wq     = (const float*)d_in[2];  // [128, 128]
    const float* bq     = (const float*)d_in[3];  // [128]
    const float* Wk     = (const float*)d_in[4];  // [128, 128]
    // d_in[5] = bk: unused (softmax is shift-invariant in Q.bk)
    const float* uin    = (const float*)d_in[6];  // [B, 2048]
    int*         out    = (int*)d_out;            // [B, 32] int32 indices
    float*       qprime = (float*)d_ws;           // [B, 128] scratch (512 KB)

    const int B = in_sizes[0] / DIMS;             // 1024

    proj_kernel<<<B / 16, 128, 0, stream>>>(target, Wq, bq, Wk, qprime);
    sample_kernel<<<B, TPB, 0, stream>>>(cand, qprime, uin, out);
}
